// EdgeMaskedMHSA_40063454937287
// MI455X (gfx1250) — compile-verified, matrix operand order checked by reading, not run on hardware
//
#include <hip/hip_runtime.h>

// ---------------------------------------------------------------------------
// EdgeMasked MHSA on gfx1250 (CDNA5, wave32, WMMA bf16 -> f32)
// B=4, N=2048, D=512, H=8, d=64
// Async global->LDS staging (double-buffered K/V tiles, X tile), packed mask.
// ---------------------------------------------------------------------------

typedef __attribute__((ext_vector_type(16))) __bf16 bf16x16;
typedef __attribute__((ext_vector_type(8)))  float  f32x8;
typedef int v4i __attribute__((ext_vector_type(4)));

#define BQ 4
#define NQ 2048
#define DQ 512
#define HQ 8
#define DH 64

typedef __attribute__((address_space(1))) v4i gv4i;   // global v4i32
typedef __attribute__((address_space(3))) v4i lv4i;   // LDS v4i32

// CDNA5 async copy: GLOBAL_LOAD_ASYNC_TO_LDS_B128 (tracked by ASYNCcnt)
__device__ __forceinline__ void async_b128(const void* g, void* l) {
    __builtin_amdgcn_global_load_async_to_lds_b128((gv4i*)g, (lv4i*)l, 0, 0);
}
__device__ __forceinline__ void wait_async0() {
    asm volatile("s_wait_asynccnt 0" ::: "memory");
}

__device__ __forceinline__ unsigned short f2bf(float f) {
    union { float f; unsigned u; } v; v.f = f;
    unsigned r = v.u + 0x7FFFu + ((v.u >> 16) & 1u);   // round-to-nearest-even
    return (unsigned short)(r >> 16);
}
__device__ __forceinline__ unsigned pack2(float lo, float hi) {
    return (unsigned)f2bf(lo) | ((unsigned)f2bf(hi) << 16);
}

union AFrag { bf16x16 v; unsigned u[8]; };

// A-layout K index for element-pair i (0..7) in half hf (lane>>4).
// ISA 7.12.2: lanes 0-15: VGPR0-3 -> K 0..7, VGPR4-7 -> K 16..23
//             lanes16-31: VGPR0-3 -> K 8..15, VGPR4-7 -> K 24..31
__device__ __forceinline__ int kidx(int i, int hf) {
    return (i < 4) ? (hf * 8 + 2 * i) : (16 + hf * 8 + 2 * (i - 4));
}

__device__ __forceinline__ bf16x16 load_frag_bf16(const unsigned short* rowptr,
                                                  int kbase, int hf) {
    AFrag f;
#pragma unroll
    for (int i = 0; i < 8; i++) {
        int k = kbase + kidx(i, hf);
        f.u[i] = *(const unsigned*)(rowptr + k);
    }
    return f.v;
}

__device__ __forceinline__ bf16x16 load_frag_f32(const float* rowptr,
                                                 int kbase, int hf) {
    AFrag f;
#pragma unroll
    for (int i = 0; i < 8; i++) {
        int k = kbase + kidx(i, hf);
        f.u[i] = pack2(rowptr[k], rowptr[k + 1]);
    }
    return f.v;
}

__device__ __forceinline__ f32x8 wmma_bf16(bf16x16 a, bf16x16 b, f32x8 c) {
    return __builtin_amdgcn_wmma_f32_16x16x32_bf16(
        false, a, false, b, (short)0, c, false, false);
}

// ---------------------------------------------------------------------------
// Prep 1: transpose + convert weights to bf16 (B^T row-major, contiguous k).
// ---------------------------------------------------------------------------
__global__ void prep_weights(const float* __restrict__ Wqkv,
                             const float* __restrict__ Wproj,
                             unsigned short* __restrict__ WqkvT,
                             unsigned short* __restrict__ WprojT) {
    int idx = blockIdx.x * blockDim.x + threadIdx.x;
    const int total1 = 1536 * 512;
    if (idx < total1) {
        int c = idx / 512, k = idx % 512;
        WqkvT[idx] = f2bf(Wqkv[k * 1536 + c]);
    } else {
        int j = idx - total1;
        if (j < 512 * 512) {
            int c = j / 512, k = j % 512;
            WprojT[j] = f2bf(Wproj[k * 512 + c]);
        }
    }
}

// ---------------------------------------------------------------------------
// Prep 2: bit-pack the adjacency mask: Mb[q*64 + w] bit j = (A[q][32w+j] > 0)
// ---------------------------------------------------------------------------
__global__ void prep_mask(const float* __restrict__ A, unsigned* __restrict__ Mb) {
    int w = blockIdx.x * blockDim.x + threadIdx.x;     // 2048*64 words
    if (w < NQ * 64) {
        int q = w >> 6, blk = w & 63;
        const float* row = A + (size_t)q * NQ + blk * 32;
        unsigned bits = 0;
#pragma unroll 8
        for (int j = 0; j < 32; j++)
            bits |= (row[j] > 0.0f ? 1u : 0u) << j;
        Mb[w] = bits;
    }
}

// ---------------------------------------------------------------------------
// QKV GEMM: block = one 16-row tile; X tile (16x512 f32, 32KB) staged in LDS
// once via async b128. Each wave computes 3 col-tiles of 64 (12 accumulators).
// Q,K -> [B,H,N,64] bf16; V -> transposed [B,H,64,N] bf16.
// ---------------------------------------------------------------------------
__global__ __launch_bounds__(256) void qkv_gemm(
    const float* __restrict__ X, const unsigned short* __restrict__ Wt,
    const float* __restrict__ bqkv,
    unsigned short* __restrict__ Qb, unsigned short* __restrict__ Kb,
    unsigned short* __restrict__ Vt) {
    __shared__ __align__(16) float Xs[16 * DQ];
    int t = threadIdx.x;
    int wslot = t >> 5, lane = t & 31;
    int row0 = blockIdx.x * 16;
    int m = lane & 15, hf = lane >> 4;

    // stage 16x512 f32 (contiguous 32KB) via async copies
    const float* src = X + (size_t)row0 * DQ;
#pragma unroll
    for (int i = 0; i < 8; i++) {
        int idx = (i * 256 + t) * 4;                   // float index, 16B chunks
        async_b128(src + idx, Xs + idx);
    }
    wait_async0();
    __syncthreads();

    f32x8 acc[3][4] = {};
    const float* xrow = Xs + m * DQ;
    for (int kb = 0; kb < DQ; kb += 32) {
        bf16x16 a = load_frag_f32(xrow, kb, hf);
#pragma unroll
        for (int j = 0; j < 3; j++) {
            int ct = wslot * 3 + j;
#pragma unroll
            for (int c = 0; c < 4; c++) {
                bf16x16 b = load_frag_bf16(
                    Wt + (size_t)(ct * 64 + 16 * c + m) * DQ, kb, hf);
                acc[j][c] = wmma_bf16(a, b, acc[j][c]);
            }
        }
    }

#pragma unroll
    for (int j = 0; j < 3; j++) {
        int ct = wslot * 3 + j;
        int mat = ct >> 3, h = ct & 7, col0 = ct * 64;
#pragma unroll
        for (int c = 0; c < 4; c++) {
            float bias = bqkv[col0 + 16 * c + m];
#pragma unroll
            for (int r = 0; r < 8; r++) {
                int grow = row0 + r + 8 * hf;          // b*2048 + n
                int n = grow & (NQ - 1), bb = grow >> 11;
                int dd = 16 * c + m;
                unsigned short bv = f2bf(acc[j][c][r] + bias);
                size_t bh = (size_t)(bb * HQ + h);
                if (mat == 0)      Qb[(bh * NQ + n) * DH + dd] = bv;
                else if (mat == 1) Kb[(bh * NQ + n) * DH + dd] = bv;
                else               Vt[(bh * DH + dd) * NQ + n] = bv;
            }
        }
    }
}

// ---------------------------------------------------------------------------
// Flash attention. Block = one (b,h) x 8 consecutive 16-query tiles (8 waves).
// Double-buffered async staging: while iteration i computes from LDS buffer
// cur, the async unit fills buffer cur^1 with the next 32-key K/V tiles.
// One wait+barrier per iteration; DMA overlaps WMMA + softmax work.
// ---------------------------------------------------------------------------
__global__ __launch_bounds__(256) void attn_kernel(
    const unsigned short* __restrict__ Qb, const unsigned short* __restrict__ Kb,
    const unsigned short* __restrict__ Vt, const unsigned* __restrict__ Mb,
    unsigned short* __restrict__ Y) {
    __shared__ __align__(16) unsigned short KsT[2][32 * DH]; // 32 keys x 64
    __shared__ __align__(16) unsigned short VsT[2][DH * 32]; // 64 feat x 32 keys
    __shared__ unsigned Pbuf32[8][256];                      // per-wave 16x32 bf16
    int t = threadIdx.x;
    int wslot = t >> 5, lane = t & 31;
    int bh = blockIdx.x >> 4;                               // 0..31 = b*8 + h
    int qt = (blockIdx.x & 15) * 8 + wslot;                 // 0..127
    int q0 = qt * 16;
    int m = lane & 15, hf = lane >> 4;

    unsigned short* P = (unsigned short*)Pbuf32[wslot];
    const unsigned short* Kbase = Kb + (size_t)bh * NQ * DH;
    const unsigned short* Vbase = Vt + (size_t)bh * DH * NQ;

    const unsigned short* Qrow = Qb + ((size_t)bh * NQ + q0 + m) * DH;
    bf16x16 aq0 = load_frag_bf16(Qrow, 0, hf);
    bf16x16 aq1 = load_frag_bf16(Qrow, 32, hf);

    float mrowv[8], lrow[8];
#pragma unroll
    for (int r = 0; r < 8; r++) { mrowv[r] = -3.0e38f; lrow[r] = 0.0f; }
    f32x8 acc[4] = {};
    const float scale = 0.125f;                             // 64^-0.5

    // staging roles (per thread, one b128 for K, one for V per tile)
    int koff = t * 8;                                       // K tile contiguous 4KB
    int vdd = t >> 2, vseg = (t & 3) * 8;                   // V rows, 16B segments

    // prefill buffer 0 with the first tile
    async_b128(Kbase + (size_t)0 * DH + koff, KsT[0] + koff);
    async_b128(Vbase + (size_t)vdd * NQ + 0 + vseg, VsT[0] + vdd * 32 + vseg);

    for (int jb = 0; jb < NQ; jb += 32) {
        int cur = (jb >> 5) & 1;
        wait_async0();                                      // own tile loads done
        __syncthreads();                                    // everyone's landed;
                                                            // prev reads of cur^1 done
        if (jb + 32 < NQ) {                                 // start next tile DMA
            int nxt = cur ^ 1;
            async_b128(Kbase + (size_t)(jb + 32) * DH + koff, KsT[nxt] + koff);
            async_b128(Vbase + (size_t)vdd * NQ + (jb + 32) + vseg,
                       VsT[nxt] + vdd * 32 + vseg);
        }

        unsigned mw[8];
#pragma unroll
        for (int r = 0; r < 8; r++)
            mw[r] = Mb[(size_t)(q0 + r + 8 * hf) * 64 + (jb >> 5)];

        float pvv[2][8], bmax[8];
#pragma unroll
        for (int r = 0; r < 8; r++) bmax[r] = -3.0e38f;

#pragma unroll
        for (int s = 0; s < 2; s++) {
            const unsigned short* Krow = KsT[cur] + (s * 16 + m) * DH;
            bf16x16 bk0 = load_frag_bf16(Krow, 0, hf);
            bf16x16 bk1 = load_frag_bf16(Krow, 32, hf);
            f32x8 S = {};
            S = wmma_bf16(aq0, bk0, S);
            S = wmma_bf16(aq1, bk1, S);
            int kl = s * 16 + m;                            // key local 0..31
            int kg = jb + kl;                               // key global
#pragma unroll
            for (int r = 0; r < 8; r++) {
                int qg = q0 + r + 8 * hf;
                float bit = ((mw[r] >> kl) & 1u) ? 1.0f : 0.0f;
                float add = (bit + (qg == kg ? 1.0f : 0.0f) - 1.0f) * 1.0e9f;
                float sv = S[r] * scale + add;
                pvv[s][r] = sv;
                bmax[r] = fmaxf(bmax[r], sv);
            }
        }

#pragma unroll
        for (int r = 0; r < 8; r++) {
            float v = bmax[r];
            for (int off = 1; off < 16; off <<= 1)
                v = fmaxf(v, __shfl_xor(v, off, 32));
            float newm = fmaxf(mrowv[r], v);
            float alpha = __expf(mrowv[r] - newm);
            mrowv[r] = newm;
            float p0 = __expf(pvv[0][r] - newm);
            float p1 = __expf(pvv[1][r] - newm);
            float psum = p0 + p1;
            for (int off = 1; off < 16; off <<= 1)
                psum += __shfl_xor(psum, off, 32);
            lrow[r] = lrow[r] * alpha + psum;
#pragma unroll
            for (int c = 0; c < 4; c++) acc[c][r] *= alpha;
            int mm = r + 8 * hf;
            P[mm * 32 + m]      = f2bf(p0);
            P[mm * 32 + 16 + m] = f2bf(p1);
        }
        __asm__ volatile("" ::: "memory");                  // LDS in-order in wave

        AFrag pf;
#pragma unroll
        for (int i = 0; i < 8; i++) {
            int k = kidx(i, hf);
            pf.u[i] = *(const unsigned*)(P + m * 32 + k);
        }
        bf16x16 ap = pf.v;

#pragma unroll
        for (int c = 0; c < 4; c++) {
            const unsigned short* Vrow = VsT[cur] + (16 * c + m) * 32;
            bf16x16 bv = load_frag_bf16(Vrow, 0, hf);
            acc[c] = wmma_bf16(ap, bv, acc[c]);
        }
        __asm__ volatile("" ::: "memory");
    }

    // normalize + scatter: Y[b][256*h + 4*dd + (q>>9)][q & 511] = O[b,h,q,dd]
    int b = bh >> 3, h = bh & 7;
#pragma unroll
    for (int r = 0; r < 8; r++) {
        float inv = 1.0f / lrow[r];
        int qg = q0 + r + 8 * hf;
#pragma unroll
        for (int c = 0; c < 4; c++) {
            int dd = 16 * c + m;
            int yrow = 256 * h + 4 * dd + (qg >> 9);
            int ycol = qg & 511;
            Y[((size_t)b * NQ + yrow) * DQ + ycol] = f2bf(acc[c][r] * inv);
        }
    }
}

// ---------------------------------------------------------------------------
// Output projection: out[8192 x 512] = Y(bf16) @ Wproj + b_proj (f32 out).
// ---------------------------------------------------------------------------
__global__ __launch_bounds__(256) void proj_gemm(
    const unsigned short* __restrict__ Y, const unsigned short* __restrict__ Wt,
    const float* __restrict__ bproj, float* __restrict__ out) {
    int wave = blockIdx.x * 8 + (threadIdx.x >> 5);
    int lane = threadIdx.x & 31;
    int rt = wave >> 3, ct = wave & 7;
    int row0 = rt * 16, col0 = ct * 64;
    int m = lane & 15, hf = lane >> 4;

    const unsigned short* yrow = Y + (size_t)(row0 + m) * DQ;
    f32x8 acc[4] = {};
    for (int kb = 0; kb < DQ; kb += 32) {
        bf16x16 a = load_frag_bf16(yrow, kb, hf);
#pragma unroll
        for (int c = 0; c < 4; c++) {
            bf16x16 b = load_frag_bf16(Wt + (size_t)(col0 + 16 * c + m) * DQ, kb, hf);
            acc[c] = wmma_bf16(a, b, acc[c]);
        }
    }
#pragma unroll
    for (int c = 0; c < 4; c++) {
        float bias = bproj[col0 + 16 * c + m];
#pragma unroll
        for (int r = 0; r < 8; r++) {
            int grow = row0 + r + 8 * hf;
            out[(size_t)grow * DQ + col0 + 16 * c + m] = acc[c][r] + bias;
        }
    }
}

// ---------------------------------------------------------------------------
extern "C" void kernel_launch(void* const* d_in, const int* in_sizes, int n_in,
                              void* d_out, int out_size, void* d_ws, size_t ws_size,
                              hipStream_t stream) {
    (void)in_sizes; (void)n_in; (void)out_size; (void)ws_size;
    const float* X     = (const float*)d_in[0];
    const float* Amask = (const float*)d_in[1];
    const float* Wqkv  = (const float*)d_in[2];
    const float* bqkv  = (const float*)d_in[3];
    const float* Wproj = (const float*)d_in[4];
    const float* bproj = (const float*)d_in[5];
    float* out = (float*)d_out;

    const size_t QKV_ELEMS = (size_t)BQ * HQ * NQ * DH;
    unsigned short* Qb     = (unsigned short*)d_ws;
    unsigned short* Kb     = Qb + QKV_ELEMS;
    unsigned short* Vt     = Kb + QKV_ELEMS;
    unsigned short* Ybuf   = Vt + QKV_ELEMS;
    unsigned short* WqkvT  = Ybuf + (size_t)8192 * 512;
    unsigned short* WprojT = WqkvT + (size_t)1536 * 512;
    unsigned*       Mbits  = (unsigned*)(WprojT + (size_t)512 * 512);

    {
        int total = 1536 * 512 + 512 * 512;
        prep_weights<<<(total + 255) / 256, 256, 0, stream>>>(Wqkv, Wproj, WqkvT, WprojT);
    }
    prep_mask<<<(NQ * 64 + 255) / 256, 256, 0, stream>>>(Amask, Mbits);
    qkv_gemm<<<512, 256, 0, stream>>>(X, WqkvT, bqkv, Qb, Kb, Vt);
    attn_kernel<<<512, 256, 0, stream>>>(Qb, Kb, Vt, Mbits, Ybuf);
    proj_gemm<<<512, 256, 0, stream>>>(Ybuf, WprojT, bproj, out);
}
